// MinCostMatcher_52218212385065
// MI455X (gfx1250) — matrix operand order repaired and checked by reading, not compiled, not run on hardware
//
#include <hip/hip_runtime.h>
#include <hip/hip_bf16.h>

typedef __attribute__((ext_vector_type(16))) _Float16 v16h;
typedef __attribute__((ext_vector_type(8)))  float    v8f;

#define ALPHA 0.25f
#define EPS   1e-8f

// Problem shapes (fixed by the reference's setup_inputs)
#define BB   8
#define WW   128
#define HH   128
#define CC   80
#define MM   100
#define NN   (WW * HH)        // 16384
#define MT   7                // ceil(100/16) m-tiles
#define NBLK 64               // n blocks
#define NB   (NN / NBLK)      // 256 anchors per block

__device__ __forceinline__ float fast_rcp(float x) {
    return __builtin_amdgcn_rcpf(x);   // v_rcp_f32, ~1 ulp: plenty for argmin
}

// ---------------------------------------------------------------------------
// Kernel 1: focal class-cost difference, f32 -> f16
//   diff[b,n,c] = ALPHA*(1-p)^2*(-log(p+eps)) - (1-ALPHA)*p^2*(-log(1-p+eps))
// ---------------------------------------------------------------------------
__global__ __launch_bounds__(256) void focal_diff_kernel(
    const float* __restrict__ cls_pred, _Float16* __restrict__ diff, int total)
{
    int i = blockIdx.x * blockDim.x + threadIdx.x;
    if (i >= total) return;
    float p   = cls_pred[i];
    float q   = 1.0f - p;
    float neg = (1.0f - ALPHA) * p * p * (-__logf(q + EPS));
    float pos = ALPHA * q * q * (-__logf(p + EPS));
    diff[i] = (_Float16)(pos - neg);
}

// ---------------------------------------------------------------------------
// Kernel 2: WMMA matcher. One wave per (n-block, m-tile, batch).
// ---------------------------------------------------------------------------
__global__ __launch_bounds__(32) void match_kernel(
    const _Float16* __restrict__ diff,      // (B,N,C) f16
    const float*    __restrict__ loc_pred,  // (B,N,4)
    const float*    __restrict__ cls_true,  // (B,M,C) one-hot
    const float*    __restrict__ loc_true,  // (B,M,4)
    float* __restrict__ pvals, int* __restrict__ pidx)
{
    const int blk  = blockIdx.x;            // n block
    const int mt   = blockIdx.y;            // m tile
    const int b    = blockIdx.z;
    const int lane = threadIdx.x;           // 0..31 (wave32)
    const int half = (lane < 16) ? 0 : 1;
    const int col  = lane & 15;
    const int m0   = mt * 16;

    // Stage the 16 ground-truth boxes of this m-tile in LDS, then pull the 8
    // rows this lane's C-fragment touches into registers (invariant over n).
    __shared__ float lds_box[16][4];
    if (lane < 16) {
        int m = m0 + lane;
        #pragma unroll
        for (int j = 0; j < 4; ++j)
            lds_box[lane][j] = (m < MM) ? loc_true[((size_t)b * MM + m) * 4 + j] : 0.0f;
    }
    __syncthreads();

    float ty1[8], tx1[8], ty2[8], tx2[8], t_area[8];
    #pragma unroll
    for (int r = 0; r < 8; ++r) {
        int mloc = half * 8 + r;
        ty1[r] = lds_box[mloc][0];
        tx1[r] = lds_box[mloc][1];
        ty2[r] = lds_box[mloc][2];
        tx2[r] = lds_box[mloc][3];
        t_area[r] = fmaxf(ty2[r] - ty1[r], 0.0f) * fmaxf(tx2[r] - tx1[r], 0.0f);
    }

    // A fragments (one-hot tile), 3 k-steps of 32 channels (c=80 padded to 96).
    // 16-bit A 16x32 layout: row = lane&15;
    //   Klocal(e) = half*16 + e      (contiguous, matches B)
    v16h a[3];
    const int mrow = m0 + col;
    #pragma unroll
    for (int s = 0; s < 3; ++s) {
        #pragma unroll
        for (int e = 0; e < 16; ++e) {
            int kl = half * 16 + e;
            int ch = 32 * s + kl;
            float v = (mrow < MM && ch < CC)
                          ? cls_true[((size_t)b * MM + mrow) * CC + ch] : 0.0f;
            a[s][e] = (_Float16)v;
        }
    }

    float minv[8];
    int   mini[8];
    #pragma unroll
    for (int r = 0; r < 8; ++r) { minv[r] = 3.4e38f; mini[r] = 0; }

    const int nbase = blk * NB;
    for (int t = 0; t < NB / 16; ++t) {
        const int n0 = nbase + t * 16;
        const int n  = n0 + col;

        // B fragments: diff row n, channels contiguous -> aligned 32B loads.
        // B k order (contiguous, same as A): lanes 0-15 hold K 0..15, lanes 16-31 K 16..31.
        const _Float16* drow = diff + ((size_t)b * NN + n) * CC;
        __builtin_prefetch(drow + 16 * CC, 0, 0);            // next n-tile
        v16h b0 = *(const v16h*)(drow + half * 16);          // K 0..31
        v16h b1 = *(const v16h*)(drow + 32 + half * 16);     // K 32..63
        v16h b2 = {};
        if (half == 0) b2 = *(const v16h*)(drow + 64);       // K 64..79 (rest pad)

        v8f cm = {};
        cm = __builtin_amdgcn_wmma_f32_16x16x32_f16(false, a[0], false, b0,
                                                    (short)0, cm, false, false);
        cm = __builtin_amdgcn_wmma_f32_16x16x32_f16(false, a[1], false, b1,
                                                    (short)0, cm, false, false);
        cm = __builtin_amdgcn_wmma_f32_16x16x32_f16(false, a[2], false, b2,
                                                    (short)0, cm, false, false);

        // Predicted box for this lane's anchor n (shared across the 8 rows).
        const float* lp = loc_pred + ((size_t)b * NN + n) * 4;
        float py1 = lp[0] * (1.0f / WW);
        float px1 = lp[1] * (1.0f / HH);
        float py2 = lp[2] * (1.0f / WW);
        float px2 = lp[3] * (1.0f / HH);
        float p_area = fmaxf(py2 - py1, 0.0f) * fmaxf(px2 - px1, 0.0f);

        #pragma unroll
        for (int r = 0; r < 8; ++r) {
            float ih = fmaxf(fminf(py2, ty2[r]) - fmaxf(py1, ty1[r]), 0.0f);
            float iw = fmaxf(fminf(px2, tx2[r]) - fmaxf(px1, tx1[r]), 0.0f);
            float inter = ih * iw;
            float uni   = p_area + t_area[r] - inter;
            // branchless: v_rcp + v_cndmask instead of exec-masked IEEE div
            float iou   = (uni > 0.0f) ? inter * fast_rcp(fmaxf(uni, EPS)) : 0.0f;
            float eh    = fmaxf(py2, ty2[r]) - fminf(py1, ty1[r]);
            float ew    = fmaxf(px2, tx2[r]) - fminf(px1, tx1[r]);
            float enc   = eh * ew;
            float pen   = (enc > 0.0f) ? (enc - uni) * fast_rcp(fmaxf(enc, EPS))
                                       : 0.0f;
            float giou  = iou - pen;
            float reg   = fabsf(ty1[r] - py1) + fabsf(tx1[r] - px1) +
                          fabsf(ty2[r] - py2) + fabsf(tx2[r] - px2);
            // total = 2*cls + 5*reg + 2*(1 - giou)
            float tot   = fmaf(2.0f, cm[r], fmaf(5.0f, reg, fmaf(-2.0f, giou, 2.0f)));
            if (tot < minv[r]) { minv[r] = tot; mini[r] = n; }  // strict: first wins
        }
    }

    // Cross-lane argmin within each 16-lane half (row m fixed, n spread on lanes).
    #pragma unroll
    for (int r = 0; r < 8; ++r) {
        float v  = minv[r];
        int   ix = mini[r];
        #pragma unroll
        for (int off = 8; off >= 1; off >>= 1) {
            float ov = __shfl_xor(v, off, 32);
            int   oi = __shfl_xor(ix, off, 32);
            if (ov < v || (ov == v && oi < ix)) { v = ov; ix = oi; }
        }
        minv[r] = v; mini[r] = ix;
    }
    if (col == 0) {
        #pragma unroll
        for (int r = 0; r < 8; ++r) {
            int mloc = half * 8 + r;
            size_t po = ((size_t)b * (MT * 16) + (m0 + mloc)) * NBLK + blk;
            pvals[po] = minv[r];
            pidx[po]  = mini[r];
        }
    }
}

// ---------------------------------------------------------------------------
// Kernel 3: fold the 64 n-block partials per (b,m); emit (b, argmin, cls_id).
// ---------------------------------------------------------------------------
__global__ __launch_bounds__(256) void reduce_kernel(
    const float* __restrict__ pvals, const int* __restrict__ pidx,
    const float* __restrict__ cls_true, int* __restrict__ out)
{
    int i = blockIdx.x * blockDim.x + threadIdx.x;  // b*M + m
    if (i >= BB * MM) return;
    int b = i / MM, m = i % MM;

    const float* pv = pvals + ((size_t)b * (MT * 16) + m) * NBLK;
    const int*   pi = pidx  + ((size_t)b * (MT * 16) + m) * NBLK;
    float best = pv[0];
    int   bidx = pi[0];
    for (int k = 1; k < NBLK; ++k) {       // ascending blocks: '<' keeps first min
        float v = pv[k];
        if (v < best || (v == best && pi[k] < bidx)) { best = v; bidx = pi[k]; }
    }

    const float* ct = cls_true + ((size_t)b * MM + m) * CC;
    float bm = ct[0];
    int   ci = 0;
    for (int k = 1; k < CC; ++k)
        if (ct[k] > bm) { bm = ct[k]; ci = k; }  // first-max (one-hot argmax)

    out[i * 3 + 0] = b;
    out[i * 3 + 1] = bidx;
    out[i * 3 + 2] = ci;
}

// ---------------------------------------------------------------------------
extern "C" void kernel_launch(void* const* d_in, const int* in_sizes, int n_in,
                              void* d_out, int out_size, void* d_ws, size_t ws_size,
                              hipStream_t stream)
{
    const float* cls_pred = (const float*)d_in[0];  // (B,W,H,C)
    const float* loc_pred = (const float*)d_in[1];  // (B,W,H,4)
    const float* cls_true = (const float*)d_in[2];  // (B,M,C)
    const float* loc_true = (const float*)d_in[3];  // (B,M,4)
    // d_in[4] = reg_mask: unused by the reference output
    int* out = (int*)d_out;                         // (B,M,3) int32

    // Workspace layout
    char* ws = (char*)d_ws;
    const size_t diff_elems = (size_t)BB * NN * CC;             // 10,485,760 f16
    _Float16* diff  = (_Float16*)ws;
    size_t off      = diff_elems * sizeof(_Float16);
    const size_t pcount = (size_t)BB * (MT * 16) * NBLK;        // 57,344
    float* pvals    = (float*)(ws + off); off += pcount * sizeof(float);
    int*   pidx     = (int*)(ws + off);

    // 1) focal class-cost difference
    int total = (int)diff_elems;
    focal_diff_kernel<<<(total + 255) / 256, 256, 0, stream>>>(cls_pred, diff, total);

    // 2) WMMA matcher: one wave per (n-block, m-tile, batch)
    match_kernel<<<dim3(NBLK, MT, BB), 32, 0, stream>>>(
        diff, loc_pred, cls_true, loc_true, pvals, pidx);

    // 3) final reduction + class id
    reduce_kernel<<<(BB * MM + 255) / 256, 256, 0, stream>>>(pvals, pidx, cls_true, out);
}
